// S2Convolution_10101763080814
// MI455X (gfx1250) — matrix operand order repaired and checked by reading, not compiled, not run on hardware
//
#include <hip/hip_runtime.h>
#include <hip/hip_bf16.h>

// ---------------------------------------------------------------------------
// S2 -> SO3 convolution for MI455X (gfx1250), wave32 + WMMA f16 (K=32 == F_IN)
// out[b,g, O2(l) + u*d + m] = (1/sqrt(32)) * sum_f psi[f,g,l^2+u] * x[b,f,l^2+m]
// psi[f,g,i] = (1/12) * sum_n Y[n,i] * w[f,g,n]
// Store-bound problem (232 MB out @ 23.3 TB/s ~ 10us floor); inputs L2-resident.
// Precision: f16 hi/lo split, 3x v_wmma_f32_16x16x32_f16, f32 accumulate.
// ---------------------------------------------------------------------------

typedef _Float16 v16h __attribute__((ext_vector_type(16)));
typedef _Float16 v8h  __attribute__((ext_vector_type(8)));
typedef float    v8f  __attribute__((ext_vector_type(8)));

#define LMAX    10
#define F_IN    32
#define F_OUT   32
#define N_GRID  144
#define BATCH   1024
#define S2_DIM  121
#define SO3_DIM 1771

// ---- workspace layout (units: _Float16 elements) --------------------------
constexpr size_t PSI_ELEMS  = (size_t)S2_DIM * F_OUT * F_IN;   // [i][g][f]
constexpr size_t XP_ELEMS   = (size_t)BATCH * S2_DIM * F_IN;   // [b][i][f]
constexpr size_t XP_PAD     = 2048;   // slack for masked tail reads (m >= d)
constexpr size_t OFF_PSI_HI = 0;
constexpr size_t OFF_PSI_LO = OFF_PSI_HI + PSI_ELEMS;
constexpr size_t OFF_XP_HI  = OFF_PSI_LO + PSI_ELEMS;
constexpr size_t OFF_XP_LO  = OFF_XP_HI + XP_ELEMS + XP_PAD;
// total ~16.4 MB of d_ws

union V16 { v16h v; v8h h[2]; };

// ---------------------------------------------------------------------------
// K1a: psi[f,g,i] = (1/12) sum_n w[f,g,n] * Y[n,i]; write hi/lo f16 in
// A-fragment-friendly layout psi_pack[i][g][f].
// ---------------------------------------------------------------------------
__global__ void psi_pack_kernel(const float* __restrict__ w,
                                const float* __restrict__ Y,
                                _Float16* __restrict__ psi_hi,
                                _Float16* __restrict__ psi_lo) {
    const int i = blockIdx.x;           // S2 index 0..120
    const int f = threadIdx.x & 31;
    const int g = threadIdx.x >> 5;
    const float* wp = w + ((size_t)f * F_OUT + g) * N_GRID;
    float s = 0.0f;
#pragma unroll 4
    for (int n = 0; n < N_GRID; ++n)
        s += wp[n] * Y[(size_t)n * S2_DIM + i];
    s *= (1.0f / 12.0f);                // 1/sqrt(N_GRID)
    const _Float16 hi = (_Float16)s;
    const _Float16 lo = (_Float16)(s - (float)hi);
    const size_t idx = ((size_t)i * F_OUT + g) * F_IN + f;
    psi_hi[idx] = hi;
    psi_lo[idx] = lo;
}

// ---------------------------------------------------------------------------
// K1b: repack x[b,f,i] (f32) -> hi/lo f16 x_pack[b][i][f] (B-fragment layout,
// f fastest so each lane's 32 K-halves are one contiguous 64B run).
// ---------------------------------------------------------------------------
__global__ void x_pack_kernel(const float* __restrict__ x,
                              _Float16* __restrict__ xp_hi,
                              _Float16* __restrict__ xp_lo) {
    const size_t t = (size_t)blockIdx.x * blockDim.x + threadIdx.x;
    const int    f = (int)(t & 31);
    const size_t r = t >> 5;
    const int    i = (int)(r % S2_DIM);
    const size_t b = r / S2_DIM;
    const float  v = x[(b * F_IN + f) * S2_DIM + i];
    const _Float16 hi = (_Float16)v;
    const _Float16 lo = (_Float16)(v - (float)hi);
    const size_t idx = (b * S2_DIM + i) * F_IN + f;
    xp_hi[idx] = hi;
    xp_lo[idx] = lo;
}

// ---------------------------------------------------------------------------
// K2: per-l WMMA kernel. GEMM per (b, u): C[g, m] = sum_f psi[f,g,O+u] *
// x[b,f,O+m].  M = g (2 gtiles of 16), N = m (MT tiles of 16), K = f = 32.
// Wave layout per 256-thread block: wave = {gtile (2)} x {b-group (4)};
// each wave owns A-fragments (psi) for its (u, gtile) and loops 8 b values.
// Lanes 0-15 of a C row hold consecutive m -> coalesced 64B store segments.
// ---------------------------------------------------------------------------
template <int L>
__global__ __launch_bounds__(256)
void s2conv_wmma_kernel(const _Float16* __restrict__ xp_hi,
                        const _Float16* __restrict__ xp_lo,
                        const _Float16* __restrict__ ps_hi,
                        const _Float16* __restrict__ ps_lo,
                        float* __restrict__ out) {
    constexpr int D  = 2 * L + 1;
    constexpr int O  = L * L;                    // S2 offset for this l
    constexpr int O2 = (4 * L * L * L - L) / 3;  // SO3 offset for this l
    constexpr int MT = (D + 15) / 16;            // m tiles (padded)

    const int lane  = threadIdx.x & 31;
    const int wave  = threadIdx.x >> 5;
    const int gtile = wave & 1;
    const int u     = blockIdx.x;                       // 0..D-1
    const int bbase = blockIdx.y * 32 + (wave >> 1) * 8;

    // ---- A fragment (16-bit A 16x32 layout, ISA 7.12.2):
    // lanes 0-15:  halves 0..7 -> K=f 0..7,  halves 8..15 -> K=f 16..23
    // lanes 16-31: halves 0..7 -> K=f 8..15, halves 8..15 -> K=f 24..31
    const int    gA    = gtile * 16 + (lane & 15);
    const int    f0a   = (lane >= 16) ? 8 : 0;
    const size_t abase = ((size_t)(O + u) * F_OUT + gA) * F_IN;
    V16 a_hi, a_lo;
    a_hi.h[0] = *(const v8h*)(ps_hi + abase + f0a);
    a_hi.h[1] = *(const v8h*)(ps_hi + abase + f0a + 16);
    a_lo.h[0] = *(const v8h*)(ps_lo + abase + f0a);
    a_lo.h[1] = *(const v8h*)(ps_lo + abase + f0a + 16);

    // ---- B fragment layout: lane&15 = N (=m), same K order as A:
    // lanes 0-15 carry K=f 0..7,16..23; lanes 16-31 carry K=f 8..15,24..31.
    const int f0b = (lane >= 16) ? 8 : 0;
    const int nB  = lane & 15;
    const float scale = 0.17677669529663688f;    // 1/sqrt(F_IN)

    for (int bi = 0; bi < 8; ++bi) {
        const int b = bbase + bi;
        if (bi < 7) { // warm L0/L2 for next b's x fragment
            __builtin_prefetch(xp_hi + ((size_t)(b + 1) * S2_DIM + O + nB) * F_IN + f0b, 0, 1);
        }
#pragma unroll
        for (int mt = 0; mt < MT; ++mt) {
            const int m0 = mt * 16 + nB;  // may exceed D-1: compute with full
            const int iB = O + m0;        // EXEC (WMMA req.), mask only stores
            const size_t bb = ((size_t)b * S2_DIM + iB) * F_IN + f0b;
            V16 b_hi, b_lo;
            b_hi.h[0] = *(const v8h*)(xp_hi + bb);
            b_hi.h[1] = *(const v8h*)(xp_hi + bb + 16);
            b_lo.h[0] = *(const v8h*)(xp_lo + bb);
            b_lo.h[1] = *(const v8h*)(xp_lo + bb + 16);

            // split-precision accumulate: hi*hi + hi*lo + lo*hi (f32 C/D)
            v8f acc = {};
            acc = __builtin_amdgcn_wmma_f32_16x16x32_f16(
                      false, a_hi.v, false, b_hi.v, (short)0, acc, false, false);
            acc = __builtin_amdgcn_wmma_f32_16x16x32_f16(
                      false, a_hi.v, false, b_lo.v, (short)0, acc, false, false);
            acc = __builtin_amdgcn_wmma_f32_16x16x32_f16(
                      false, a_lo.v, false, b_hi.v, (short)0, acc, false, false);

            if (m0 < D) {
                // C layout: VGPR r -> row M=r (lanes 0-15) / M=r+8 (16-31)
                const int gBase = gtile * 16 + ((lane >= 16) ? 8 : 0);
                float* op = out + ((size_t)b * F_OUT + gBase) * SO3_DIM
                                + (size_t)(O2 + u * D + m0);
#pragma unroll
                for (int r = 0; r < 8; ++r)
                    op[(size_t)r * SO3_DIM] = acc[r] * scale;
            }
        }
    }
}

// ---------------------------------------------------------------------------
extern "C" void kernel_launch(void* const* d_in, const int* in_sizes, int n_in,
                              void* d_out, int out_size, void* d_ws, size_t ws_size,
                              hipStream_t stream) {
    const float* x = (const float*)d_in[0];   // (1024, 32, 121)
    const float* w = (const float*)d_in[1];   // (32, 32, 144)
    const float* Y = (const float*)d_in[2];   // (144, 121)
    float* out = (float*)d_out;               // (1024, 32, 1771)

    _Float16* wsh   = (_Float16*)d_ws;        // ~16.4 MB used
    _Float16* ps_hi = wsh + OFF_PSI_HI;
    _Float16* ps_lo = wsh + OFF_PSI_LO;
    _Float16* xp_hi = wsh + OFF_XP_HI;
    _Float16* xp_lo = wsh + OFF_XP_LO;

    // K1a: psi pack (121 blocks x 1024 threads; one thread per (i,g,f))
    psi_pack_kernel<<<S2_DIM, 1024, 0, stream>>>(w, Y, ps_hi, ps_lo);

    // K1b: x pack (one thread per element; 3,964,928 / 256 = 15,488 blocks)
    x_pack_kernel<<<(BATCH * S2_DIM * F_IN) / 256, 256, 0, stream>>>(x, xp_hi, xp_lo);

    // K2: one launch per l; grid = (u slots, 32 b-chunks), 256 threads.
#define L_LAUNCH(Lv)                                                          \
    s2conv_wmma_kernel<Lv><<<dim3(2 * (Lv) + 1, 32), 256, 0, stream>>>(       \
        xp_hi, xp_lo, ps_hi, ps_lo, out)
    L_LAUNCH(0);  L_LAUNCH(1);  L_LAUNCH(2);  L_LAUNCH(3);
    L_LAUNCH(4);  L_LAUNCH(5);  L_LAUNCH(6);  L_LAUNCH(7);
    L_LAUNCH(8);  L_LAUNCH(9);  L_LAUNCH(10);
#undef L_LAUNCH
}
